// NeuralSplineFlow_24532853194801
// MI455X (gfx1250) — matrix operand order repaired and checked by reading, not compiled, not run on hardware
//
#include <hip/hip_runtime.h>
#include <hip/hip_bf16.h>

typedef __attribute__((ext_vector_type(16))) _Float16 v16h;
typedef __attribute__((ext_vector_type(8)))  _Float16 v8h;
typedef __attribute__((ext_vector_type(8)))  float    v8f;

#define BATCH   65536
#define FDIM    16
#define CDIM    128
#define HDIM    256
#define LNUM    8
#define KBINS   8
#define TFEAT   8
#define PDIM    23
#define NP3     184
#define TBND    3.0f
#define MINW    0.001f
#define MIND    0.001f

#define TILE_M  64
#define SA      264      // activation LDS row stride (halves): 528B -> 4-bank rotation
#define SPAN    40       // weight panel LDS row stride (halves): 80B rows
#define K1PAD   160      // GEMM1 K padded 136 -> 160 (5 K-steps of 32)
#define SP      188      // param LDS row stride (floats)

#define OFF_P0   0
#define OFF_P1   20480
#define OFF_ACTA 40960
#define OFF_ACTB (OFF_ACTA + TILE_M*SA*2)
#define OFF_PBUF (OFF_ACTB + TILE_M*SA*2)
#define SMEM_BYTES (OFF_PBUF + TILE_M*SP*4)

#define USE_TDM 1

#if USE_TDM
typedef unsigned int u32x4 __attribute__((ext_vector_type(4)));
typedef int          i32x4 __attribute__((ext_vector_type(4)));
typedef int          i32x8 __attribute__((ext_vector_type(8)));

// Issue one TDM 2D tile load: nrows x 64B from global (row stride kpad_halves*2 B)
// into LDS at lds_off with 80B row stride (pad 16B after every 64B).
// D# encoding per CDNA5 ISA 8.3/8.4.
__device__ __forceinline__ void tdm_panel_load(const _Float16* gsrc,
                                               unsigned lds_off,
                                               int nrows, int kpad_halves)
{
  const unsigned long long ga = (unsigned long long)gsrc;
  u32x4 g0;
  g0.x = 1u;                                            // count=1, user desc
  g0.y = lds_off;                                       // lds_addr (bytes)
  g0.z = (unsigned)(ga & 0xffffffffull);                // global_addr[31:0]
  g0.w = ((unsigned)(ga >> 32) & 0x01ffffffu)           // global_addr[56:32]
         | 0x80000000u;                                 // type=2 ("image")
  const int td0 = kpad_halves >> 1;                     // tensor_dim0 (4B units)
  i32x8 g1;
  g1[0] = (2 << 16)        // data_size = 4B
        | (1 << 20)        // pad_enable
        | (3 << 22)        // pad_interval: 16 dwords (64B)
        | (3 << 25);       // pad_amount:   4 dwords (16B)
  g1[1] = (td0 & 0xffff) << 16;                         // tensor_dim0[15:0]
  g1[2] = ((td0 >> 16) & 0xffff) | ((nrows & 0xffff) << 16); // dim0 hi | dim1 lo
  g1[3] = ((nrows >> 16) & 0xffff) | (16 << 16);        // dim1 hi | tile_dim0=16
  g1[4] = (nrows & 0xffff);                             // tile_dim1 = nrows
  g1[5] = td0;                                          // tensor_dim0_stride lo32
  g1[6] = 0;                                            // stride hi | dim1_stride lo
  g1[7] = 0;
  const i32x4 z4 = {0, 0, 0, 0};
#if __clang_major__ >= 23
  const i32x8 z8 = {0, 0, 0, 0, 0, 0, 0, 0};
  __builtin_amdgcn_tensor_load_to_lds(g0, g1, z4, z4, z8, 0);
#else
  __builtin_amdgcn_tensor_load_to_lds(g0, g1, z4, z4, 0);
#endif
}
#endif

__device__ __forceinline__ float softplusf(float v) {
  return (v > 20.f) ? v : log1pf(expf(v));
}

// ---------------------------------------------------------------------------
// Block-cooperative GEMM: A [64 x K] f16 in LDS, B streamed from global
// ([N][Kpad] f16, N-major) through a double-buffered LDS K-panel (TDM DMA),
// C in f32 WMMA accumulators. Wave w: M-tile = w&3, N-half = w>>2.
// ---------------------------------------------------------------------------
template<int NT, int KS, bool RELU, bool PB>
__device__ void gemm_block(const _Float16* __restrict__ Wp, int Kpad,
                           const float* __restrict__ bias,
                           int a_off, int o_off, int ncols)
{
  extern __shared__ char smem[];
  const int tid  = threadIdx.x;
  const int lane = tid & 31;
  const int wave = tid >> 5;
  const int rlo  = lane & 15;   // row (A) / col (B/C) within 16-tile
  const int qq   = lane >> 4;   // lane quadrant selects K / M halves
  const int mt   = wave & 3;    // M tile (0..3)
  const int nb   = (wave >> 2) * (NT / 2); // first N tile for this wave
  const int N    = NT * 16;

  _Float16* p0 = (_Float16*)(smem + OFF_P0);
  _Float16* p1 = (_Float16*)(smem + OFF_P1);
  const _Float16* actA = (const _Float16*)(smem + a_off);

  // stage K-panel 0
#if USE_TDM
  if (wave == 0) {
    tdm_panel_load(Wp, OFF_P0, N, Kpad);
    __builtin_amdgcn_s_wait_tensorcnt(0);
  }
#else
  if (tid < N) {
    const uint4* src = (const uint4*)(Wp + (size_t)tid * Kpad);
    uint4* dst = (uint4*)(p0 + tid * SPAN);
    dst[0] = src[0]; dst[1] = src[1]; dst[2] = src[2]; dst[3] = src[3];
  }
#endif
  __syncthreads();

  const v8f vzero = {0.f,0.f,0.f,0.f,0.f,0.f,0.f,0.f};
  v8f acc[NT / 2];
#pragma unroll
  for (int i = 0; i < NT / 2; ++i) acc[i] = vzero;

  for (int ks = 0; ks < KS; ++ks) {
    _Float16* cur = (ks & 1) ? p1 : p0;
    const bool dopf = (ks + 1 < KS);

    // start DMA of next K-panel (overlaps WMMA below)
#if USE_TDM
    if (dopf && wave == 0)
      tdm_panel_load(Wp + (ks + 1) * 32, (ks & 1) ? OFF_P0 : OFF_P1, N, Kpad);
#else
    _Float16* nxt = (ks & 1) ? p0 : p1;
    uint4 pf0, pf1, pf2, pf3;
    if (dopf && tid < N) {
      const uint4* src = (const uint4*)(Wp + (size_t)tid * Kpad + (ks + 1) * 32);
      pf0 = src[0]; pf1 = src[1]; pf2 = src[2]; pf3 = src[3];
    }
#endif

    // A fragment: lane holds row (mt*16+rlo); halves {K0..7,K16..23} (q=0)
    // or {K8..15,K24..31} (q=1) -> two 16B LDS reads
    const _Float16* ap = actA + (size_t)(mt * 16 + rlo) * SA + ks * 32 + qq * 8;
    v8h alo = *(const v8h*)ap;
    v8h ahi = *(const v8h*)(ap + 16);
    v16h afrag = __builtin_shufflevector(alo, ahi,
        0,1,2,3,4,5,6,7,8,9,10,11,12,13,14,15);

    // preload ALL B fragments first so LDS latency overlaps across tiles
    v16h bfr[NT / 2];
#pragma unroll
    for (int nt = 0; nt < NT / 2; ++nt) {
      const _Float16* bp = cur + (size_t)((nb + nt) * 16 + rlo) * SPAN + qq * 8;
      v8h blo = *(const v8h*)bp;
      v8h bhi = *(const v8h*)(bp + 16);
      bfr[nt] = __builtin_shufflevector(blo, bhi,
          0,1,2,3,4,5,6,7,8,9,10,11,12,13,14,15);
    }
#pragma unroll
    for (int nt = 0; nt < NT / 2; ++nt) {
      acc[nt] = __builtin_amdgcn_wmma_f32_16x16x32_f16(
          false, afrag, false, bfr[nt], (short)0, acc[nt], false, false);
    }

#if USE_TDM
    if (dopf && wave == 0)
      __builtin_amdgcn_s_wait_tensorcnt(0);
#else
    if (dopf && tid < N) {
      uint4* dst = (uint4*)(nxt + tid * SPAN);
      dst[0] = pf0; dst[1] = pf1; dst[2] = pf2; dst[3] = pf3;
    }
#endif
    __syncthreads();
  }

  // epilogue: C element (m = qq*8+j, n = rlo) per lane
  if (!PB) {
    _Float16* out = (_Float16*)(smem + o_off);
#pragma unroll
    for (int nt = 0; nt < NT / 2; ++nt) {
      const int col = (nb + nt) * 16 + rlo;
      const float bv = bias[col];
#pragma unroll
      for (int j = 0; j < 8; ++j) {
        float v = acc[nt][j] + bv;
        if (RELU) v = fmaxf(v, 0.f);
        out[(size_t)(mt * 16 + qq * 8 + j) * SA + col] = (_Float16)v;
      }
    }
  } else {
    float* pb = (float*)(smem + OFF_PBUF);
#pragma unroll
    for (int nt = 0; nt < NT / 2; ++nt) {
      const int col = (nb + nt) * 16 + rlo;
      if (col < ncols) {
        const float bv = bias[col];
#pragma unroll
        for (int j = 0; j < 8; ++j) {
          pb[(size_t)(mt * 16 + qq * 8 + j) * SP + col] = acc[nt][j] + bv;
        }
      }
    }
  }
  __syncthreads();
}

// ---------------------------------------------------------------------------
// One coupling layer, fully fused: build A1 = [xi | ctx | 0], 3 GEMMs
// (f16 WMMA, f32 accum), RQ spline + interleave + permutation + logdet.
// ---------------------------------------------------------------------------
__global__ __launch_bounds__(256, 1) void nsf_layer(
    const float* __restrict__ x_in, float* __restrict__ x_out,
    const _Float16* __restrict__ ctx16,
    const _Float16* __restrict__ W1p, const _Float16* __restrict__ W2p,
    const _Float16* __restrict__ W3p,
    const float* __restrict__ b1, const float* __restrict__ b2,
    const float* __restrict__ b3,
    const int* __restrict__ perm, float* __restrict__ logdet, int parity)
{
  extern __shared__ char smem[];
  const int tid  = threadIdx.x;
  const int base = blockIdx.x * TILE_M;

  // build A1 tile [64 x 160] f16: cols 0..7 = xi, 8..135 = ctx, 136..159 = 0
  {
    const int r  = tid >> 2;
    const int qp = tid & 3;
    const int s  = base + r;
    _Float16* a1 = (_Float16*)(smem + OFF_ACTA) + (size_t)r * SA;
    const _Float16* cr = ctx16 + (size_t)s * CDIM;
    const int c0 = qp * 40, c1 = c0 + 40;
    for (int c = c0; c < c1; ++c) {
      _Float16 v;
      if (c < 8)        v = (_Float16)x_in[s * FDIM + 2 * c + parity];
      else if (c < 136) v = cr[c - 8];
      else              v = (_Float16)0.f;
      a1[c] = v;
    }
  }
  __syncthreads();

  gemm_block<16, 5, true,  false>(W1p, K1PAD, b1, OFF_ACTA, OFF_ACTB, HDIM);
  gemm_block<16, 8, true,  false>(W2p, HDIM,  b2, OFF_ACTB, OFF_ACTA, HDIM);
  gemm_block<12, 8, false, true >(W3p, HDIM,  b3, OFF_ACTA, 0,        NP3);

  // spline: one thread per sample, 8 features each
  if (tid < TILE_M) {
    const int s = base + tid;
    float* pr = (float*)(smem + OFF_PBUF) + (size_t)tid * SP;
    float ldsum = 0.f;
    float ynew[TFEAT], xid[TFEAT];
    for (int t = 0; t < TFEAT; ++t) {
      const float* pp = pr + t * PDIM;
      const float xv  = x_in[s * FDIM + 2 * t + (1 - parity)]; // transformed feat
      xid[t]          = x_in[s * FDIM + 2 * t + parity];       // identity feat

      float mw = -1e30f, mh = -1e30f;
      for (int j = 0; j < KBINS; ++j) {
        mw = fmaxf(mw, pp[j]);
        mh = fmaxf(mh, pp[KBINS + j]);
      }
      float ew[KBINS], eh[KBINS], sw = 0.f, sh = 0.f;
      for (int j = 0; j < KBINS; ++j) {
        ew[j] = expf(pp[j] - mw);          sw += ew[j];
        eh[j] = expf(pp[KBINS + j] - mh);  sh += eh[j];
      }
      float cw[KBINS + 1], ch[KBINS + 1];
      cw[0] = -TBND; ch[0] = -TBND;
      float aw = 0.f, ah = 0.f;
      const float rw = (1.f - MINW * KBINS) / sw;
      const float rh = (1.f - MINW * KBINS) / sh;
      for (int j = 0; j < KBINS; ++j) {
        aw += MINW + rw * ew[j];
        ah += MINW + rh * eh[j];
        cw[j + 1] = -TBND + 2.f * TBND * aw;
        ch[j + 1] = -TBND + 2.f * TBND * ah;
      }
      cw[KBINS] = TBND; ch[KBINS] = TBND;
      float dd[KBINS + 1];
      dd[0] = 1.f; dd[KBINS] = 1.f;  // softplus(log(expm1(1-MIN_D)))+MIN_D == 1
      for (int j = 1; j < KBINS; ++j)
        dd[j] = MIND + softplusf(pp[2 * KBINS + j - 1]);

      const bool inside = (xv >= -TBND) && (xv <= TBND);
      const float xs = fminf(fmaxf(xv, -TBND), TBND);
      int b = 0;
      for (int j = 1; j < KBINS; ++j) if (xs >= cw[j]) b = j;

      const float x0 = cw[b], bw = cw[b + 1] - cw[b];
      const float y0 = ch[b], bh = ch[b + 1] - ch[b];
      const float delta = bh / bw;
      const float d0 = dd[b], d1 = dd[b + 1];
      const float th = (xs - x0) / bw;
      const float th1m = th * (1.f - th);
      const float num = bh * (delta * th * th + d0 * th1m);
      const float den = delta + (d0 + d1 - 2.f * delta) * th1m;
      const float y = y0 + num / den;
      const float dnum = delta * delta *
          (d1 * th * th + 2.f * delta * th1m + d0 * (1.f - th) * (1.f - th));
      if (inside) ldsum += logf(dnum) - 2.f * logf(den);
      ynew[t] = inside ? y : xv;
    }

    // interleave (inv of feature split) then apply permutation; reuse pr row
    for (int k = 0; k < TFEAT; ++k) {
      if (parity == 0) { pr[2 * k] = xid[k];  pr[2 * k + 1] = ynew[k]; }
      else             { pr[2 * k] = ynew[k]; pr[2 * k + 1] = xid[k];  }
    }
    if (perm) {
      for (int j = 0; j < FDIM; ++j)
        x_out[(size_t)s * FDIM + j] = pr[perm[j]];
    } else {
      for (int j = 0; j < FDIM; ++j)
        x_out[(size_t)s * FDIM + j] = pr[j];
    }
    logdet[s] += ldsum;
  }
}

// ---------------------------------------------------------------------------
// Prep: pack weights f32 -> f16, transposed to [N][Kpad] (K contiguous).
// ---------------------------------------------------------------------------
__global__ void pack_weights(const float* __restrict__ W1,
                             const float* __restrict__ W2,
                             const float* __restrict__ W3,
                             _Float16* __restrict__ W1p,
                             _Float16* __restrict__ W2p,
                             _Float16* __restrict__ W3p)
{
  const int idx = blockIdx.x * blockDim.x + threadIdx.x;
  if (idx < LNUM * HDIM * K1PAD) {
    const int l = idx / (HDIM * K1PAD);
    const int r = idx % (HDIM * K1PAD);
    const int n = r / K1PAD, k = r % K1PAD;
    const float v = (k < 136) ? W1[((size_t)l * 136 + k) * HDIM + n] : 0.f;
    W1p[idx] = (_Float16)v;
  }
  if (idx < LNUM * HDIM * HDIM) {
    const int l = idx / (HDIM * HDIM);
    const int r = idx % (HDIM * HDIM);
    const int n = r / HDIM, k = r % HDIM;
    W2p[idx] = (_Float16)W2[((size_t)l * HDIM + k) * HDIM + n];
  }
  if (idx < LNUM * 192 * HDIM) {
    const int l = idx / (192 * HDIM);
    const int r = idx % (192 * HDIM);
    const int n = r / HDIM, k = r % HDIM;
    const float v = (n < NP3) ? W3[((size_t)l * HDIM + k) * NP3 + n] : 0.f;
    W3p[idx] = (_Float16)v;
  }
}

__global__ void prep_xc(const float* __restrict__ inputs,
                        const float* __restrict__ context,
                        _Float16* __restrict__ ctx16,
                        float* __restrict__ x0,
                        float* __restrict__ logdet)
{
  const int i = blockIdx.x * blockDim.x + threadIdx.x;
  if (i < BATCH * CDIM) ctx16[i] = (_Float16)context[i];
  if (i < BATCH * FDIM) x0[i] = fminf(fmaxf(inputs[i], -1.f), 1.f);
  if (i < BATCH)        logdet[i] = 0.f;
}

__global__ void finalize(const float* __restrict__ x,
                         const float* __restrict__ logdet,
                         float* __restrict__ out)
{
  const int i = blockIdx.x * blockDim.x + threadIdx.x;
  if (i < BATCH * FDIM) out[i] = fminf(fmaxf(x[i], -1.f), 1.f);
  if (i < BATCH)        out[(size_t)BATCH * FDIM + i] = logdet[i];
}

// ---------------------------------------------------------------------------
extern "C" void kernel_launch(void* const* d_in, const int* in_sizes, int n_in,
                              void* d_out, int out_size, void* d_ws, size_t ws_size,
                              hipStream_t stream)
{
  const float* inputs  = (const float*)d_in[0];
  const float* context = (const float*)d_in[1];
  const float* W1      = (const float*)d_in[2];
  const float* b1      = (const float*)d_in[3];
  const float* W2      = (const float*)d_in[4];
  const float* b2      = (const float*)d_in[5];
  const float* W3      = (const float*)d_in[6];
  const float* b3      = (const float*)d_in[7];
  const int*   perms   = (const int*)d_in[8];

  char* ws = (char*)d_ws;
  size_t o = 0;
  _Float16* ctx16 = (_Float16*)(ws + o); o += (size_t)BATCH * CDIM * 2;
  float*    xA    = (float*)(ws + o);    o += (size_t)BATCH * FDIM * 4;
  float*    xB    = (float*)(ws + o);    o += (size_t)BATCH * FDIM * 4;
  float*    ld    = (float*)(ws + o);    o += (size_t)BATCH * 4;
  _Float16* W1p   = (_Float16*)(ws + o); o += (size_t)LNUM * HDIM * K1PAD * 2;
  _Float16* W2p   = (_Float16*)(ws + o); o += (size_t)LNUM * HDIM * HDIM * 2;
  _Float16* W3p   = (_Float16*)(ws + o); o += (size_t)LNUM * 192 * HDIM * 2;
  (void)o; (void)ws_size; (void)in_sizes; (void)n_in; (void)out_size;

  pack_weights<<<(LNUM * HDIM * HDIM + 255) / 256, 256, 0, stream>>>(
      W1, W2, W3, W1p, W2p, W3p);
  prep_xc<<<(BATCH * CDIM + 255) / 256, 256, 0, stream>>>(
      inputs, context, ctx16, xA, ld);

  float* xi_ = xA;
  float* xo_ = xB;
  for (int l = 0; l < LNUM; ++l) {
    const int* pm = (l < LNUM - 1) ? (perms + l * FDIM) : nullptr;
    nsf_layer<<<BATCH / TILE_M, 256, SMEM_BYTES, stream>>>(
        xi_, xo_, ctx16,
        W1p + (size_t)l * HDIM * K1PAD,
        W2p + (size_t)l * HDIM * HDIM,
        W3p + (size_t)l * 192 * HDIM,
        b1 + l * HDIM, b2 + l * HDIM, b3 + l * NP3,
        pm, ld, l & 1);
    float* t = xi_; xi_ = xo_; xo_ = t;
  }

  finalize<<<(BATCH * FDIM + 255) / 256, 256, 0, stream>>>(
      xi_, ld, (float*)d_out);
}
